// Corm_23175643529490
// MI455X (gfx1250) — matrix operand order repaired and checked by reading, not compiled, not run on hardware
//
#include <hip/hip_runtime.h>
#include <hip/hip_bf16.h>

typedef __attribute__((ext_vector_type(16))) __bf16 v16bf;
typedef __attribute__((ext_vector_type(8)))  float  v8f;

#define HEADS 32
#define HD    128
#define QLEN  2048
#define KVLEN 2048
#define SCALE 0.08838834764831845f   /* 1/sqrt(128) */
#define NEG_BIG (-1.0e30f)

#define K_STRIDE  136   /* bf16 elems per K LDS row (272B = 68 dwords, bank-rotated) */
#define VT_STRIDE 36    /* bf16 elems per VT LDS row (72B = 18 dwords, bank-rotated) */

#if defined(__has_builtin)
#if __has_builtin(__builtin_amdgcn_s_wait_dscnt)
#define WAIT_DS() __builtin_amdgcn_s_wait_dscnt(0)
#endif
#endif
#ifndef WAIT_DS
#define WAIT_DS() asm volatile("s_wait_dscnt 0" ::: "memory")
#endif

// ---- LDS gathers into WMMA B-operand layout ----
__device__ __forceinline__ v16bf lds_bk(const __bf16* p) {   // 16B aligned
    union { v16bf v; uint4 q[2]; } u;
    u.q[0] = *(const uint4*)p;
    u.q[1] = *(const uint4*)(p + 16);
    return u.v;
}
__device__ __forceinline__ v16bf lds_bv(const __bf16* p) {   // 8B aligned
    union { v16bf v; uint2 d[4]; } u;
    u.d[0] = *(const uint2*)(p);
    u.d[1] = *(const uint2*)(p + 4);
    u.d[2] = *(const uint2*)(p + 16);
    u.d[3] = *(const uint2*)(p + 20);
    return u.v;
}

// A-matrix (16x32 bf16) per-lane load from f32: elements 0..7 <- p[0..7],
// 8..15 <- p[16..23] (caller pre-offsets by 8*hf).
__device__ __forceinline__ v16bf load_a_bf16(const float* __restrict__ p) {
    const float4* p4 = (const float4*)p;
    float4 t0 = p4[0], t1 = p4[1], t2 = p4[4], t3 = p4[5];
    v16bf r;
    r[0]=(__bf16)t0.x; r[1]=(__bf16)t0.y; r[2]=(__bf16)t0.z; r[3]=(__bf16)t0.w;
    r[4]=(__bf16)t1.x; r[5]=(__bf16)t1.y; r[6]=(__bf16)t1.z; r[7]=(__bf16)t1.w;
    r[8]=(__bf16)t2.x; r[9]=(__bf16)t2.y; r[10]=(__bf16)t2.z; r[11]=(__bf16)t2.w;
    r[12]=(__bf16)t3.x; r[13]=(__bf16)t3.y; r[14]=(__bf16)t3.z; r[15]=(__bf16)t3.w;
    return r;
}

__global__ void init_corm_kernel(unsigned* __restrict__ c, int nwords) {
    int i = blockIdx.x * blockDim.x + threadIdx.x;
    if (i < nwords) c[i] = 0u;
}

__global__ __launch_bounds__(256)
void corm_flash_attn(const float* __restrict__ qg,
                     const float* __restrict__ kg,
                     const float* __restrict__ vg,
                     const float* __restrict__ cmask,
                     float* __restrict__ out,
                     unsigned* __restrict__ corm)
{
    const int t    = threadIdx.x;
    const int lane = t & 31;
    const int wave = t >> 5;             // 0..7
    const int n    = lane & 15;
    const int hf   = lane >> 4;
    const int q0b  = blockIdx.x * 128;   // block q range
    const int q0   = q0b + wave * 16;    // this wave's 16 q rows
    const int bh   = blockIdx.y;
    const int b    = bh >> 5;
    const int head = bh & 31;

    const float thresh = cmask[0];

    const size_t rstride = (size_t)HEADS * HD;   // 4096 floats
    const float* qbase = qg + ((size_t)b * QLEN)  * rstride + (size_t)head * HD;
    const float* kbase = kg + ((size_t)b * KVLEN) * rstride + (size_t)head * HD;
    const float* vbase = vg + ((size_t)b * KVLEN) * rstride + (size_t)head * HD;

    __shared__ __align__(16) __bf16 kshare[32 * K_STRIDE];    // K tile, row-major
    __shared__ __align__(16) __bf16 vtshare[HD * VT_STRIDE];  // V tile, transposed
    __shared__ __align__(16) float  pshare[8 * 512];          // per-wave P staging
    float* pwave = &pshare[wave * 512];

    // staging thread mapping
    const int kr  = t >> 3;          // K: row 0..31
    const int kc0 = (t & 7) * 16;    // K: col base
    const int vr0 = (t >> 5) * 4;    // V: row base 0..28
    const int vc0 = (t & 31) * 4;    // V: col base 0..124

    // ---- Load this wave's Q tile (16x128) to bf16 A-layout ----
    v16bf Aq[4];
    {
        const float* qrow = qbase + (size_t)(q0 + n) * rstride + 8 * hf;
        #pragma unroll
        for (int kc = 0; kc < 4; ++kc)
            Aq[kc] = load_a_bf16(qrow + kc * 32);
    }

    v8f Oacc[8];
    #pragma unroll
    for (int s = 0; s < 8; ++s) Oacc[s] = (v8f)0.0f;
    float mrow[8], lrow[8];
    #pragma unroll
    for (int r = 0; r < 8; ++r) { mrow[r] = NEG_BIG; lrow[r] = 0.0f; }

    const int kvend_blk = q0b + 128;     // block causal extent
    const int kvend_w   = q0 + 16;       // wave causal extent

    // ================= pass 1: flash attention =================
    for (int kv0 = 0; kv0 < kvend_blk; kv0 += 32) {
        // ---- cooperative stage: K (row-major bf16) + V (transposed bf16) ----
        {
            const float4* ks = (const float4*)(kbase + (size_t)(kv0 + kr) * rstride + kc0);
            float4 a0 = ks[0], a1 = ks[1], a2 = ks[2], a3 = ks[3];
            union { __bf16 h[16]; uint4 q[2]; } pk;
            pk.h[0]=(__bf16)a0.x; pk.h[1]=(__bf16)a0.y; pk.h[2]=(__bf16)a0.z; pk.h[3]=(__bf16)a0.w;
            pk.h[4]=(__bf16)a1.x; pk.h[5]=(__bf16)a1.y; pk.h[6]=(__bf16)a1.z; pk.h[7]=(__bf16)a1.w;
            pk.h[8]=(__bf16)a2.x; pk.h[9]=(__bf16)a2.y; pk.h[10]=(__bf16)a2.z; pk.h[11]=(__bf16)a2.w;
            pk.h[12]=(__bf16)a3.x; pk.h[13]=(__bf16)a3.y; pk.h[14]=(__bf16)a3.z; pk.h[15]=(__bf16)a3.w;
            uint4* kd = (uint4*)&kshare[kr * K_STRIDE + kc0];
            kd[0] = pk.q[0]; kd[1] = pk.q[1];

            const float* vs = vbase + (size_t)(kv0 + vr0) * rstride + vc0;
            float4 r0 = *(const float4*)(vs);
            float4 r1 = *(const float4*)(vs + rstride);
            float4 r2 = *(const float4*)(vs + 2 * rstride);
            float4 r3 = *(const float4*)(vs + 3 * rstride);
            union { __bf16 h[4]; uint2 d; } u;
            u.h[0]=(__bf16)r0.x; u.h[1]=(__bf16)r1.x; u.h[2]=(__bf16)r2.x; u.h[3]=(__bf16)r3.x;
            *(uint2*)&vtshare[(vc0 + 0) * VT_STRIDE + vr0] = u.d;
            u.h[0]=(__bf16)r0.y; u.h[1]=(__bf16)r1.y; u.h[2]=(__bf16)r2.y; u.h[3]=(__bf16)r3.y;
            *(uint2*)&vtshare[(vc0 + 1) * VT_STRIDE + vr0] = u.d;
            u.h[0]=(__bf16)r0.z; u.h[1]=(__bf16)r1.z; u.h[2]=(__bf16)r2.z; u.h[3]=(__bf16)r3.z;
            *(uint2*)&vtshare[(vc0 + 2) * VT_STRIDE + vr0] = u.d;
            u.h[0]=(__bf16)r0.w; u.h[1]=(__bf16)r1.w; u.h[2]=(__bf16)r2.w; u.h[3]=(__bf16)r3.w;
            *(uint2*)&vtshare[(vc0 + 3) * VT_STRIDE + vr0] = u.d;
        }
        __syncthreads();

        if (kv0 < kvend_w) {             // wave-uniform guard: EXEC stays full
            v8f S[2];
            #pragma unroll
            for (int nt = 0; nt < 2; ++nt) {
                v8f s = (v8f)0.0f;
                const __bf16* krow = &kshare[(nt * 16 + n) * K_STRIDE + 8 * hf];
                #pragma unroll
                for (int kc = 0; kc < 4; ++kc) {
                    v16bf Bk = lds_bk(krow + kc * 32);
                    s = __builtin_amdgcn_wmma_f32_16x16x32_bf16(
                            false, Aq[kc], false, Bk, (short)0, s, false, false);
                }
                S[nt] = s;
            }

            #pragma unroll
            for (int nt = 0; nt < 2; ++nt) {
                const int key = kv0 + nt * 16 + n;
                #pragma unroll
                for (int r = 0; r < 8; ++r) {
                    const int qr = q0 + r + 8 * hf;
                    float sv = S[nt][r] * SCALE;
                    S[nt][r] = (key > qr) ? NEG_BIG : sv;
                }
            }

            #pragma unroll
            for (int r = 0; r < 8; ++r) {
                float tm = fmaxf(S[0][r], S[1][r]);
                tm = fmaxf(tm, __shfl_xor(tm, 1, 32));
                tm = fmaxf(tm, __shfl_xor(tm, 2, 32));
                tm = fmaxf(tm, __shfl_xor(tm, 4, 32));
                tm = fmaxf(tm, __shfl_xor(tm, 8, 32));
                const float mnew  = fmaxf(mrow[r], tm);
                const float alpha = __expf(mrow[r] - mnew);
                mrow[r] = mnew;
                const float p0 = __expf(S[0][r] - mnew);
                const float p1 = __expf(S[1][r] - mnew);
                S[0][r] = p0; S[1][r] = p1;
                float rs = p0 + p1;
                rs += __shfl_xor(rs, 1, 32);
                rs += __shfl_xor(rs, 2, 32);
                rs += __shfl_xor(rs, 4, 32);
                rs += __shfl_xor(rs, 8, 32);
                lrow[r] = lrow[r] * alpha + rs;
                #pragma unroll
                for (int sl = 0; sl < 8; ++sl) Oacc[sl][r] *= alpha;
            }

            // P: C-layout -> wave-private LDS -> bf16 A-layout
            #pragma unroll
            for (int nt = 0; nt < 2; ++nt)
                #pragma unroll
                for (int r = 0; r < 8; ++r)
                    pwave[(r + 8 * hf) * 32 + nt * 16 + n] = S[nt][r];
            WAIT_DS();
            v16bf Ap = load_a_bf16(&pwave[n * 32 + 8 * hf]);

            // O += P * V : V B-operand straight from transposed LDS (no cvt)
            #pragma unroll
            for (int sl = 0; sl < 8; ++sl) {
                v16bf Bv = lds_bv(&vtshare[(sl * 16 + n) * VT_STRIDE + 8 * hf]);
                Oacc[sl] = __builtin_amdgcn_wmma_f32_16x16x32_bf16(
                               false, Ap, false, Bv, (short)0, Oacc[sl], false, false);
            }
        }
        __syncthreads();
    }

    // ---- normalize + store O ----
    float invl[8];
    #pragma unroll
    for (int r = 0; r < 8; ++r) invl[r] = 1.0f / lrow[r];
    float* obase = out + (((size_t)b * QLEN + q0) * HEADS + head) * HD;
    #pragma unroll
    for (int sl = 0; sl < 8; ++sl)
        #pragma unroll
        for (int r = 0; r < 8; ++r) {
            const int m = r + 8 * hf;
            obase[(size_t)m * rstride + sl * 16 + n] = Oacc[sl][r] * invl[r];
        }

    // ================= pass 2: CORM flags =================
    // prob(q,k) >= t  <=>  s(q,k) >= log(t) + log(l_final) + m_final
    float thr[8];
    const float lt = __logf(thresh);
    #pragma unroll
    for (int r = 0; r < 8; ++r) thr[r] = lt + __logf(lrow[r]) + mrow[r];

    unsigned* cbase = corm + ((size_t)b * HEADS + head) * KVLEN;
    for (int kv0 = 0; kv0 < kvend_blk; kv0 += 32) {
        {   // stage K only
            const float4* ks = (const float4*)(kbase + (size_t)(kv0 + kr) * rstride + kc0);
            float4 a0 = ks[0], a1 = ks[1], a2 = ks[2], a3 = ks[3];
            union { __bf16 h[16]; uint4 q[2]; } pk;
            pk.h[0]=(__bf16)a0.x; pk.h[1]=(__bf16)a0.y; pk.h[2]=(__bf16)a0.z; pk.h[3]=(__bf16)a0.w;
            pk.h[4]=(__bf16)a1.x; pk.h[5]=(__bf16)a1.y; pk.h[6]=(__bf16)a1.z; pk.h[7]=(__bf16)a1.w;
            pk.h[8]=(__bf16)a2.x; pk.h[9]=(__bf16)a2.y; pk.h[10]=(__bf16)a2.z; pk.h[11]=(__bf16)a2.w;
            pk.h[12]=(__bf16)a3.x; pk.h[13]=(__bf16)a3.y; pk.h[14]=(__bf16)a3.z; pk.h[15]=(__bf16)a3.w;
            uint4* kd = (uint4*)&kshare[kr * K_STRIDE + kc0];
            kd[0] = pk.q[0]; kd[1] = pk.q[1];
        }
        __syncthreads();

        if (kv0 < kvend_w) {
            #pragma unroll
            for (int nt = 0; nt < 2; ++nt) {
                v8f s = (v8f)0.0f;
                const __bf16* krow = &kshare[(nt * 16 + n) * K_STRIDE + 8 * hf];
                #pragma unroll
                for (int kc = 0; kc < 4; ++kc) {
                    v16bf Bk = lds_bk(krow + kc * 32);
                    s = __builtin_amdgcn_wmma_f32_16x16x32_bf16(
                            false, Aq[kc], false, Bk, (short)0, s, false, false);
                }
                const int key = kv0 + nt * 16 + n;
                unsigned f = 0u;
                #pragma unroll
                for (int r = 0; r < 8; ++r) {
                    const int qr = q0 + r + 8 * hf;
                    const float sv = s[r] * SCALE;
                    f |= (key <= qr && sv >= thr[r]) ? 1u : 0u;
                }
                f |= __shfl_xor(f, 16, 32);
                if (hf == 0 && f)
                    atomicOr(&cbase[key], 0x3F800000u);   // bits of 1.0f
            }
        }
        __syncthreads();
    }
}

extern "C" void kernel_launch(void* const* d_in, const int* in_sizes, int n_in,
                              void* d_out, int out_size, void* d_ws, size_t ws_size,
                              hipStream_t stream) {
    const float* q  = (const float*)d_in[0];
    const float* k  = (const float*)d_in[1];
    const float* v  = (const float*)d_in[2];
    const float* cm = (const float*)d_in[3];
    float* out = (float*)d_out;

    const int B = in_sizes[0] / (QLEN * HEADS * HD);

    const size_t out_elems = (size_t)B * QLEN * HEADS * HD;
    unsigned* corm = (unsigned*)(out + out_elems);
    const int nwords = B * HEADS * KVLEN;

    init_corm_kernel<<<(nwords + 255) / 256, 256, 0, stream>>>(corm, nwords);

    dim3 grid(QLEN / 128, B * HEADS);
    corm_flash_attn<<<grid, 256, 0, stream>>>(q, k, v, cm, out, corm);
}
